// LIFBenchmark_68135361183760
// MI455X (gfx1250) — matrix operand order repaired and checked by reading, compiled, not run on hardware
//
#include <hip/hip_runtime.h>

// ---------------------------------------------------------------------------
// SNN LIF benchmark, MI455X (gfx1250, wave32, WMMA)
// Phase 1: X[t*B+b, g] = sum_f spikes[t,b,f] * W[g,f]   (fp16 WMMA, f32 acc)
// Phase 2: CUBA-LIF scan over t per neuron, transposed store to [B,F,T]
// ---------------------------------------------------------------------------

typedef __attribute__((ext_vector_type(16))) _Float16 v16h;
typedef __attribute__((ext_vector_type(8)))  _Float16 v8h;
typedef __attribute__((ext_vector_type(4)))  _Float16 v4h;
typedef __attribute__((ext_vector_type(8)))  float    v8f;
typedef __attribute__((ext_vector_type(4)))  float    f32x4;

#define T_STEPS 256
#define BATCH   32
#define FDIM    2048
#define M_TOT   (T_STEPS * BATCH)   // 8192
#define N_TOT   FDIM                // 2048
#define K_TOT   FDIM                // 2048

#define BM 128
#define BN 128
#define BK 64
#define LDK (BK + 8)                // padded half-row: 144 B -> conflict-free b128

__global__ __launch_bounds__(256) void snn_gemm_f16(
    const float* __restrict__ A,    // [M_TOT, K_TOT]  spikes, row = t*B+b
    const float* __restrict__ W,    // [N_TOT, K_TOT]  weight [out, in]
    float* __restrict__ X)          // [M_TOT, N_TOT]
{
    __shared__ _Float16 As[2][BM][LDK];
    __shared__ _Float16 Bs[2][BN][LDK];

    const int tid   = threadIdx.x;
    const int wave  = tid >> 5;
    const int lane  = tid & 31;
    const int wm    = wave >> 1;        // 0..3  (M groups of 32)
    const int wn    = wave & 1;         // 0..1  (N groups of 64)
    const int row16 = lane & 15;
    const int hi    = lane >> 4;        // 0/1  (half-wave)

    const int mBase = blockIdx.y * BM;
    const int nBase = blockIdx.x * BN;

    // global->LDS staging assignment (same pattern for A and B tiles)
    const int lr = tid >> 4;            // base row 0..15
    const int lk = (tid & 15) << 2;     // float4 column 0,4,...,60

    v8f acc[2][4];
    #pragma unroll
    for (int i = 0; i < 2; ++i)
        #pragma unroll
        for (int j = 0; j < 4; ++j)
            acc[i][j] = {};

    auto load_stage = [&](int stage, int buf) {
        const int kBase = stage * BK;
        #pragma unroll
        for (int p = 0; p < 8; ++p) {
            const int r = lr + (p << 4);
            f32x4 av = *(const f32x4*)(A + (size_t)(mBase + r) * K_TOT + kBase + lk);
            f32x4 bv = *(const f32x4*)(W + (size_t)(nBase + r) * K_TOT + kBase + lk);
            v4h ha = { (_Float16)av[0], (_Float16)av[1], (_Float16)av[2], (_Float16)av[3] };
            v4h hb = { (_Float16)bv[0], (_Float16)bv[1], (_Float16)bv[2], (_Float16)bv[3] };
            *(v4h*)(&As[buf][r][lk]) = ha;
            *(v4h*)(&Bs[buf][r][lk]) = hb;
        }
    };

    auto compute_stage = [&](int buf) {
        #pragma unroll
        for (int ks = 0; ks < 2; ++ks) {
            const int k0 = ks * 32;
            v16h afrag[2], bfrag[4];
            // A fragment (ISA 16-bit A 16x32 layout):
            //   halves 0..7  -> K = k0 + hi*8 + e
            //   halves 8..15 -> K = k0 + 16 + hi*8 + e
            #pragma unroll
            for (int tm = 0; tm < 2; ++tm) {
                const _Float16* pa = &As[buf][wm * 32 + tm * 16 + row16][k0 + hi * 8];
                v8h lo = *(const v8h*)pa;
                v8h hh = *(const v8h*)(pa + 16);
                v16h a;
                #pragma unroll
                for (int e = 0; e < 8; ++e) { a[e] = lo[e]; a[e + 8] = hh[e]; }
                afrag[tm] = a;
            }
            // B fragment: column N = row16, K = k0 + hi*8 + e (0..7), k0 + 16 + hi*8 + e (8..15)
            #pragma unroll
            for (int tn = 0; tn < 4; ++tn) {
                const _Float16* pb = &Bs[buf][wn * 64 + tn * 16 + row16][k0 + hi * 8];
                v8h lo = *(const v8h*)pb;
                v8h hh = *(const v8h*)(pb + 16);
                v16h b;
                #pragma unroll
                for (int e = 0; e < 8; ++e) { b[e] = lo[e]; b[e + 8] = hh[e]; }
                bfrag[tn] = b;
            }
            #pragma unroll
            for (int tm = 0; tm < 2; ++tm)
                #pragma unroll
                for (int tn = 0; tn < 4; ++tn)
                    acc[tm][tn] = __builtin_amdgcn_wmma_f32_16x16x32_f16(
                        false, afrag[tm], false, bfrag[tn],
                        (short)0, acc[tm][tn], false, false);
        }
    };

    load_stage(0, 0);
    __syncthreads();

    const int NSTAGES = K_TOT / BK;     // 32
    for (int s = 0; s < NSTAGES; ++s) {
        const int cur = s & 1;
        if (s + 1 < NSTAGES) load_stage(s + 1, cur ^ 1);
        compute_stage(cur);
        __syncthreads();
    }

    // C/D layout: VGPR r -> row (hi*8 + r), col = row16
    #pragma unroll
    for (int tm = 0; tm < 2; ++tm) {
        const int mrow0 = mBase + wm * 32 + tm * 16 + hi * 8;
        #pragma unroll
        for (int tn = 0; tn < 4; ++tn) {
            const int col = nBase + wn * 64 + tn * 16 + row16;
            #pragma unroll
            for (int r = 0; r < 8; ++r)
                X[(size_t)(mrow0 + r) * N_TOT + col] = acc[tm][tn][r];
        }
    }
}

// ---------------------------------------------------------------------------
// Phase 2: LIF recurrence. One thread per neuron (b, f). Reads X[t][b][f]
// coalesced across f; stages 32-step spike chunks in LDS and flushes the
// [B,F,T] layout with 128 B-contiguous wave32 stores.
// ---------------------------------------------------------------------------
#define TC 32

__global__ __launch_bounds__(256) void lif_scan(
    const float* __restrict__ X,    // [T][B][F]
    float* __restrict__ out)        // [B][F][T]
{
    __shared__ float sbuf[256][TC + 1];

    const int tid = threadIdx.x;
    const int b   = blockIdx.x >> 3;            // 32 batches * 8 f-blocks
    const int f0  = (blockIdx.x & 7) * 256;
    const int f   = f0 + tid;

    float cur = 0.f, volt = 0.f;
    const float* xp = X + (size_t)b * FDIM + f;

    for (int tc = 0; tc < T_STEPS; tc += TC) {
        #pragma unroll
        for (int t = 0; t < TC; ++t) {
            float xv = xp[(size_t)(tc + t) * (BATCH * FDIM)];
            cur  = 0.1f * cur + xv;             // i = (1-0.9)*i + x
            volt = 0.1f * volt + cur;           // v = (1-0.9)*v + i
            float s = (volt >= 1.0f) ? 1.0f : 0.0f;
            volt *= (1.0f - s);                 // hard reset
            sbuf[tid][t] = s;
        }
        __syncthreads();
        // transposed flush: wave w handles rows j = w, w+8, ... ; lane = time
        const int tt = tid & (TC - 1);
        const int j0 = tid >> 5;
        #pragma unroll
        for (int p = 0; p < 32; ++p) {
            const int j = j0 + p * 8;
            out[((size_t)b * FDIM + f0 + j) * T_STEPS + tc + tt] = sbuf[j][tt];
        }
        __syncthreads();
    }
}

// ---------------------------------------------------------------------------
extern "C" void kernel_launch(void* const* d_in, const int* in_sizes, int n_in,
                              void* d_out, int out_size, void* d_ws, size_t ws_size,
                              hipStream_t stream) {
    const float* spikes = (const float*)d_in[0];   // [T,B,F] fp32
    const float* W      = (const float*)d_in[1];   // [F,F]   fp32
    float* X   = (float*)d_ws;                     // [T*B, F] fp32 (64 MB)
    float* out = (float*)d_out;                    // [B,F,T] fp32

    dim3 gemmGrid(N_TOT / BN, M_TOT / BM);         // 16 x 64 = 1024 blocks
    snn_gemm_f16<<<gemmGrid, 256, 0, stream>>>(spikes, W, X);

    lif_scan<<<dim3((BATCH * FDIM) / 256), 256, 0, stream>>>(X, out);
}
